// TransformerBlock_7782480740618
// MI455X (gfx1250) — matrix operand order repaired and checked by reading, not compiled, not run on hardware
//
#include <hip/hip_runtime.h>
#include <hip/hip_bf16.h>

// ---------------------------------------------------------------------------
// CDNA5 (gfx1250) implementation of the 2-layer graph-transformer reference.
// All heavy GEMMs + attention run on v_wmma_f32_16x16x32_f16 (wave32 WMMA).
// Weights are pre-converted to f16 once per layer so the GEMM inner loop is
// pure WMMA + b128 loads (no v_cvt in the hot path). K is a compile-time 256
// so the 8 A fragments stay resident in VGPRs (no scratch spill) and are
// amortized over a 128-wide column group per wave.
// ---------------------------------------------------------------------------

#define L0D       256
#define L1D       128
#define NHEAD     8
#define HEAD_DIM  32
#define RBFD      16
#define NGRAPH    256
#define MAXM      128
#define DEG       16

typedef __attribute__((ext_vector_type(16))) _Float16 v16h;
typedef __attribute__((ext_vector_type(8)))  float    v8f;

// ---------------- WMMA fragment loaders (contiguous k order) ----------------
// A (16x32 f16, MxK): lane L holds row m=L%16; element j holds K = 16*(L/16) + j.
__device__ __forceinline__ v16h frag_a_f16(const _Float16* A, int lda, int row0,
                                           int k0, int lane, int nrows) {
  int m = row0 + (lane & 15);
  if (m >= nrows) m = nrows - 1;            // clamp; stores are guarded
  int kb = (lane >> 4) << 4;                // 0 or 16
  const _Float16* ar = A + (size_t)m * lda + k0;
  v16h f;
#pragma unroll
  for (int j = 0; j < 16; ++j) {
    int v = j >> 1, hl = j & 1;
    int k = kb + j; (void)v; (void)hl;
    f[j] = ar[k];
  }
  return f;
}

// A fragment sourced from f32 LDS (attention probabilities), ld = MAXM.
__device__ __forceinline__ v16h frag_a_f32(const float* S, int row0, int k0,
                                           int lane) {
  int m = row0 + (lane & 15);
  int kb = (lane >> 4) << 4;
  const float* sr = S + m * MAXM + k0;
  v16h f;
#pragma unroll
  for (int j = 0; j < 16; ++j) {
    int v = j >> 1, hl = j & 1;
    int k = kb + j; (void)v; (void)hl;
    f[j] = (_Float16)sr[k];
  }
  return f;
}

// B (32x16): lanes 0-15 hold K=0..15 (n = lane), lanes 16-31 hold K=16..31.
// Source is row-major f16 weight W16 (Kout x K): B[k][n] = W16[n0+n][k0+k].
// 16 contiguous f16 per lane -> two b128 loads, no conversion.
__device__ __forceinline__ v16h frag_b_w16(const _Float16* W16, int ldw, int n0,
                                           int k0, int lane) {
  int n  = n0 + (lane & 15);
  int kb = (lane >> 4) << 4;
  const _Float16* wr = W16 + (size_t)n * ldw + k0 + kb;
  v16h f;
#pragma unroll
  for (int j = 0; j < 16; ++j) f[j] = wr[j];
  return f;
}

// B from row-major f16 matrix (K x Nc): B[k][n] = Bm[(k0+k)*ldb + n0+n].
__device__ __forceinline__ v16h frag_b_f16(const _Float16* Bm, int ldb, int k0,
                                           int n0, int lane) {
  int n  = n0 + (lane & 15);
  int kb = k0 + ((lane >> 4) << 4);
  v16h f;
#pragma unroll
  for (int j = 0; j < 16; ++j) f[j] = Bm[(size_t)(kb + j) * ldb + n];
  return f;
}

// B for Q@K^T: B[k][n] = Kmat[(n0+n)*32 + k]  (keys row-major M x 32).
__device__ __forceinline__ v16h frag_b_kT(const _Float16* Km, int n0, int lane) {
  int n  = n0 + (lane & 15);
  int kb = (lane >> 4) << 4;
  const _Float16* kr = Km + n * HEAD_DIM + kb;
  v16h f;
#pragma unroll
  for (int j = 0; j < 16; ++j) f[j] = kr[j];
  return f;
}

// --------------------------- small utility kernels ---------------------------
__global__ __launch_bounds__(256) void offsets_kernel(const int* __restrict__ eidx,
                                                      int N, int* __restrict__ offsets) {
  __shared__ int cnt[NGRAPH];
  int tid = threadIdx.x;
  cnt[tid] = 0;
  __syncthreads();
  for (int i = tid; i < N; i += 256) atomicAdd(&cnt[eidx[i]], 1);
  __syncthreads();
  if (tid == 0) {
    int acc = 0;
    for (int g = 0; g < NGRAPH; ++g) { offsets[g] = acc; acc += cnt[g]; }
    offsets[NGRAPH] = acc;
  }
}

__global__ __launch_bounds__(256) void copy_feats_kernel(const float* __restrict__ src,
                                                         float* __restrict__ dst, int N) {
  int i = blockIdx.x * 256 + threadIdx.x;
  int total = N * L0D;
  if (i < total) {
    int row = i >> 8, c = i & 255;
    dst[i] = src[(size_t)row * (L0D + L1D) + c];
  }
}

// f32 -> f16 bulk conversion (weights, once per layer; out of the hot loops).
__global__ __launch_bounds__(256) void cvt_f16_kernel(const float* __restrict__ src,
                                                      _Float16* __restrict__ dst, int n) {
  int i = blockIdx.x * 256 + threadIdx.x;
  if (i < n) dst[i] = (_Float16)src[i];
}

// LayerNorm(256) -> f16 output. One wave per node.
__global__ __launch_bounds__(256) void layernorm_kernel(const float* __restrict__ x,
                                                        const float* __restrict__ g,
                                                        const float* __restrict__ b,
                                                        _Float16* __restrict__ out, int N) {
  int lane = threadIdx.x & 31, warp = threadIdx.x >> 5;
  int node = blockIdx.x * 8 + warp;
  if (node >= N) return;
  const float* xr = x + (size_t)node * L0D;
  float v[8], s = 0.f;
#pragma unroll
  for (int i = 0; i < 8; ++i) { v[i] = xr[lane + 32 * i]; s += v[i]; }
#pragma unroll
  for (int o = 16; o > 0; o >>= 1) s += __shfl_xor(s, o, 32);
  float mean = s * (1.f / 256.f);
  float vs = 0.f;
#pragma unroll
  for (int i = 0; i < 8; ++i) { float d = v[i] - mean; vs += d * d; }
#pragma unroll
  for (int o = 16; o > 0; o >>= 1) vs += __shfl_xor(vs, o, 32);
  float inv = rsqrtf(vs * (1.f / 256.f) + 1e-5f);
  _Float16* orow = out + (size_t)node * L0D;
#pragma unroll
  for (int i = 0; i < 8; ++i) {
    int c = lane + 32 * i;
    orow[c] = (_Float16)((v[i] - mean) * inv * g[c] + b[c]);
  }
}

// ------------------------------- WMMA GEMM ----------------------------------
// out(N x Kout) = A16(N x 256) @ W16(Kout x 256)^T + bias [+ resid].
// K == 256 at compile time: the 8 A fragments (64 VGPRs) are loaded once per
// row tile and reused across a 128-wide column group; fully unrolled inner
// loop is pure v_wmma + b128 B-tile loads (no scratch, no cvt).
__global__ __launch_bounds__(256) void gemm_wmma_kernel(const _Float16* __restrict__ A,
                                                        const _Float16* __restrict__ W16,
                                                        const float* __restrict__ bias,
                                                        const float* __restrict__ resid,
                                                        float* __restrict__ outF,
                                                        _Float16* __restrict__ outH,
                                                        int N, int Kout) {
  const int K = L0D;                       // compile-time 256
  int lane = threadIdx.x & 31, warp = threadIdx.x >> 5;
  int ntRow  = (N + 15) >> 4;
  int ntColG = Kout >> 7;                  // groups of 8 col tiles (128 cols)
  int ntasks = ntRow * ntColG;
  int stride = gridDim.x * 8;
  for (int t = blockIdx.x * 8 + warp; t < ntasks; t += stride) {
    int rt = t / ntColG, cg = t - rt * ntColG;
    int row0 = rt << 4;
    v16h a0 = frag_a_f16(A, K, row0, 0 * 32, lane, N);
    v16h a1 = frag_a_f16(A, K, row0, 1 * 32, lane, N);
    v16h a2 = frag_a_f16(A, K, row0, 2 * 32, lane, N);
    v16h a3 = frag_a_f16(A, K, row0, 3 * 32, lane, N);
    v16h a4 = frag_a_f16(A, K, row0, 4 * 32, lane, N);
    v16h a5 = frag_a_f16(A, K, row0, 5 * 32, lane, N);
    v16h a6 = frag_a_f16(A, K, row0, 6 * 32, lane, N);
    v16h a7 = frag_a_f16(A, K, row0, 7 * 32, lane, N);
#pragma unroll 1
    for (int cc = 0; cc < 8; ++cc) {
      int col0 = ((cg << 3) + cc) << 4;
      if (cc < 7)                           // hint next B tile into cache
        __builtin_prefetch(W16 + (size_t)(col0 + 16) * K, 0, 0);
      v8f c = {};
      c = __builtin_amdgcn_wmma_f32_16x16x32_f16(false, a0, false,
            frag_b_w16(W16, K, col0, 0 * 32, lane), (short)0, c, false, false);
      c = __builtin_amdgcn_wmma_f32_16x16x32_f16(false, a1, false,
            frag_b_w16(W16, K, col0, 1 * 32, lane), (short)0, c, false, false);
      c = __builtin_amdgcn_wmma_f32_16x16x32_f16(false, a2, false,
            frag_b_w16(W16, K, col0, 2 * 32, lane), (short)0, c, false, false);
      c = __builtin_amdgcn_wmma_f32_16x16x32_f16(false, a3, false,
            frag_b_w16(W16, K, col0, 3 * 32, lane), (short)0, c, false, false);
      c = __builtin_amdgcn_wmma_f32_16x16x32_f16(false, a4, false,
            frag_b_w16(W16, K, col0, 4 * 32, lane), (short)0, c, false, false);
      c = __builtin_amdgcn_wmma_f32_16x16x32_f16(false, a5, false,
            frag_b_w16(W16, K, col0, 5 * 32, lane), (short)0, c, false, false);
      c = __builtin_amdgcn_wmma_f32_16x16x32_f16(false, a6, false,
            frag_b_w16(W16, K, col0, 6 * 32, lane), (short)0, c, false, false);
      c = __builtin_amdgcn_wmma_f32_16x16x32_f16(false, a7, false,
            frag_b_w16(W16, K, col0, 7 * 32, lane), (short)0, c, false, false);
      int col = col0 + (lane & 15);
      float bv = bias ? bias[col] : 0.f;
#pragma unroll
      for (int r = 0; r < 8; ++r) {
        int row = row0 + ((lane >> 4) << 3) + r;
        if (row < N) {
          size_t o = (size_t)row * Kout + col;
          float val = c[r] + bv;
          if (resid) val += resid[o];
          if (outF) outF[o] = val;
          if (outH) outH[o] = (_Float16)val;
        }
      }
    }
  }
}

// ------------------------------ edge-gate MLP -------------------------------
__global__ __launch_bounds__(256) void edge_mlp_kernel(const float* __restrict__ rbf,
                                                       const float* __restrict__ bn_g,
                                                       const float* __restrict__ bn_b,
                                                       const float* __restrict__ Wb1,
                                                       const float* __restrict__ Wb2,
                                                       const float* __restrict__ bb2,
                                                       float* __restrict__ egate, int E) {
  __shared__ float sW1[64 * RBFD];
  __shared__ float sW2[NHEAD * 64];
  __shared__ float sg[RBFD], sb[RBFD], sb2[NHEAD];
  int tid = threadIdx.x;
  for (int i = tid; i < 64 * RBFD; i += 256) sW1[i] = Wb1[i];
  for (int i = tid; i < NHEAD * 64; i += 256) sW2[i] = Wb2[i];
  if (tid < RBFD) { sg[tid] = bn_g[tid]; sb[tid] = bn_b[tid]; }
  if (tid < NHEAD) sb2[tid] = bb2[tid];
  __syncthreads();
  for (int e = blockIdx.x * 256 + tid; e < E; e += gridDim.x * 256) {
    const float* xr = rbf + (size_t)e * RBFD;
    float x[RBFD], s = 0.f;
#pragma unroll
    for (int d = 0; d < RBFD; ++d) { x[d] = xr[d]; s += x[d]; }
    float mean = s * (1.f / RBFD);
    float vs = 0.f;
#pragma unroll
    for (int d = 0; d < RBFD; ++d) { float dd = x[d] - mean; vs += dd * dd; }
    float inv = rsqrtf(vs * (1.f / RBFD) + 1e-5f);
#pragma unroll
    for (int d = 0; d < RBFD; ++d) x[d] = (x[d] - mean) * inv * sg[d] + sb[d];
    float o[NHEAD];
#pragma unroll
    for (int k = 0; k < NHEAD; ++k) o[k] = 0.f;
    for (int j = 0; j < 64; ++j) {
      float h = 0.f;
#pragma unroll
      for (int d = 0; d < RBFD; ++d) h += sW1[j * RBFD + d] * x[d];
      h = h / (1.f + __expf(-h));           // silu
#pragma unroll
      for (int k = 0; k < NHEAD; ++k) o[k] += sW2[k * 64 + j] * h;
    }
    float* er = egate + (size_t)e * NHEAD;
#pragma unroll
    for (int k = 0; k < NHEAD; ++k)
      er[k] = 1.f / (1.f + __expf(-(o[k] + sb2[k])));   // sigmoid
  }
}

// -------------------------- attention (per graph/head) ----------------------
// Block = (graph b, head h). LDS: K/Q/V f16 tiles (zero-padded to 128x32) + a
// 128x128 f32 score matrix. S = (Q K^T)/sqrt(d) + edge bias; masked softmax is
// equivalent to restricting to rows/cols < count[b]; then O = A V via WMMA.
#define ATTN_SMEM (3 * MAXM * HEAD_DIM * 2 + MAXM * MAXM * 4)
__global__ __launch_bounds__(256) void attn_kernel(const _Float16* __restrict__ kqv16,
                                                   const float* __restrict__ egate,
                                                   const int* __restrict__ edge_index,
                                                   int E,
                                                   const int* __restrict__ offsets,
                                                   _Float16* __restrict__ o16) {
  extern __shared__ char smem[];
  _Float16* k16 = (_Float16*)smem;
  _Float16* q16 = k16 + MAXM * HEAD_DIM;
  _Float16* v16 = q16 + MAXM * HEAD_DIM;
  float*    S   = (float*)(v16 + MAXM * HEAD_DIM);

  int b = blockIdx.x >> 3, h = blockIdx.x & 7;
  int off = offsets[b];
  int Mb  = offsets[b + 1] - off;
  int tid = threadIdx.x, lane = tid & 31, warp = tid >> 5;

  for (int i = tid; i < MAXM * HEAD_DIM; i += 256) {
    k16[i] = (_Float16)0.f; q16[i] = (_Float16)0.f; v16[i] = (_Float16)0.f;
  }
  __syncthreads();
  const float inv_scale = 0.17677669529663687f;    // 1/sqrt(32)
  for (int i = tid; i < Mb * HEAD_DIM; i += 256) {
    int row = i >> 5, d = i & 31;
    size_t base = (size_t)(off + row) * (3 * L0D) + h * HEAD_DIM + d;
    k16[i] = kqv16[base];                                        // k chunk
    q16[i] = (_Float16)((float)kqv16[base + L0D] * inv_scale);   // q chunk
    v16[i] = kqv16[base + 2 * L0D];                              // v chunk
  }
  __syncthreads();

  int nt = (Mb + 15) >> 4;
  // S = Q K^T  (one K=32 WMMA per 16x16 tile: head_dim == 32)
  for (int t = warp; t < nt * nt; t += 8) {
    int it = t / nt, jt = t - it * nt;
    v16h a  = frag_a_f16(q16, HEAD_DIM, it * 16, 0, lane, MAXM);
    v16h bf = frag_b_kT(k16, jt * 16, lane);
    v8f c = {};
    c = __builtin_amdgcn_wmma_f32_16x16x32_f16(false, a, false, bf, (short)0,
                                               c, false, false);
    int col = jt * 16 + (lane & 15);
#pragma unroll
    for (int r = 0; r < 8; ++r)
      S[(it * 16 + ((lane >> 4) << 3) + r) * MAXM + col] = c[r];
  }
  __syncthreads();

  // Edge bias scatter: edges of graph b are contiguous, (src,tgt) unique.
  int estart = off * DEG, ecnt = Mb * DEG;
  for (int e = tid; e < ecnt; e += 256) {
    int ge = estart + e;
    int ls = edge_index[ge] - off;
    int lt = edge_index[E + ge] - off;
    S[ls * MAXM + lt] += egate[(size_t)ge * NHEAD + h];
  }
  __syncthreads();

  // Row softmax over valid keys; zero the 32-aligned pad so A@V sees zeros.
  if (tid < Mb) {
    float* row = S + tid * MAXM;
    float mx = -1e30f;
    for (int j = 0; j < Mb; ++j) mx = fmaxf(mx, row[j]);
    float sum = 0.f;
    for (int j = 0; j < Mb; ++j) { float ev = __expf(row[j] - mx); row[j] = ev; sum += ev; }
    float invs = 1.f / sum;
    for (int j = 0; j < Mb; ++j) row[j] *= invs;
    int c32 = (Mb + 31) & ~31;
    for (int j = Mb; j < c32; ++j) row[j] = 0.f;
  }
  __syncthreads();

  // O = A V   (out tiles: nt row-tiles x 2 column halves of head_dim)
  for (int t = warp; t < nt * 2; t += 8) {
    int it = t >> 1, n0 = (t & 1) << 4;
    v8f c = {};
    for (int j0 = 0; j0 < Mb; j0 += 32) {
      v16h a  = frag_a_f32(S, it * 16, j0, lane);
      v16h bf = frag_b_f16(v16, HEAD_DIM, j0, n0, lane);
      c = __builtin_amdgcn_wmma_f32_16x16x32_f16(false, a, false, bf, (short)0,
                                                 c, false, false);
    }
    int col = n0 + (lane & 15);
#pragma unroll
    for (int r = 0; r < 8; ++r) {
      int row = it * 16 + ((lane >> 4) << 3) + r;
      if (row < Mb)
        o16[(size_t)(off + row) * L0D + h * HEAD_DIM + col] = (_Float16)c[r];
    }
  }
}

// ------------------------- fused FF head (WMMA + reduce) --------------------
// out[n] = sum_j silu(x[n,:]·W1[j,:]) * W2[j] + b2. Block = 16-row strip; A
// fragments cached in VGPRs; silu*W2 reduced through LDS float atomics.
__global__ __launch_bounds__(256) void ff_kernel(const _Float16* __restrict__ xln,
                                                 const _Float16* __restrict__ W116,
                                                 const float* __restrict__ W2,
                                                 const float* __restrict__ b2,
                                                 float* __restrict__ out, int N) {
  __shared__ float acc[16];
  int tid = threadIdx.x, lane = tid & 31, warp = tid >> 5;
  if (tid < 16) acc[tid] = 0.f;
  __syncthreads();
  int row0 = blockIdx.x * 16;
  v16h a[8];
#pragma unroll
  for (int kk = 0; kk < 8; ++kk)
    a[kk] = frag_a_f16(xln, L0D, row0, kk * 32, lane, N);
#pragma unroll 1
  for (int jt = warp; jt < 64; jt += 8) {                 // 1024/16 col tiles
    v8f c = {};
#pragma unroll
    for (int kk = 0; kk < 8; ++kk) {
      v16h bf = frag_b_w16(W116, L0D, jt * 16, kk * 32, lane);
      c = __builtin_amdgcn_wmma_f32_16x16x32_f16(false, a[kk], false, bf,
                                                 (short)0, c, false, false);
    }
    float w2v = W2[jt * 16 + (lane & 15)];
#pragma unroll
    for (int r = 0; r < 8; ++r) {
      int row = ((lane >> 4) << 3) + r;
      float hval = c[r];
      hval = hval / (1.f + __expf(-hval));                // silu
      atomicAdd(&acc[row], hval * w2v);
    }
  }
  __syncthreads();
  if (tid < 16) {
    int node = row0 + tid;
    if (node < N) out[node] = acc[tid] + b2[0];
  }
}

// --------------------------------- launcher ---------------------------------
extern "C" void kernel_launch(void* const* d_in, const int* in_sizes, int n_in,
                              void* d_out, int out_size, void* d_ws, size_t ws_size,
                              hipStream_t stream) {
  (void)n_in; (void)out_size; (void)ws_size;
  const int N = in_sizes[0] / (L0D + L1D);
  const int E = in_sizes[1] / RBFD;

  const float* features = (const float*)d_in[0];
  const float* rbf      = (const float*)d_in[1];
  // params flatten in dict order: l1(11), l2(11), ff(5)
  auto gf = [&](int i) { return (const float*)d_in[i]; };
  const int* eidx  = (const int*)d_in[29];         // ensemble_index
  const int* edges = (const int*)d_in[30];         // edge_index (2,E)
  float* out = (float*)d_out;

  // workspace partition (256B aligned)
  char* w = (char*)d_ws;
  auto alloc = [&](size_t bytes) {
    char* p = w;
    w += (bytes + 255) & ~(size_t)255;
    return p;
  };
  int*       offsets = (int*)alloc((NGRAPH + 1) * sizeof(int));
  float*     feats   = (float*)alloc((size_t)N * L0D * sizeof(float));
  _Float16*  xln     = (_Float16*)alloc((size_t)N * L0D * sizeof(_Float16));
  _Float16*  kqv16   = (_Float16*)alloc((size_t)N * 3 * L0D * sizeof(_Float16));
  _Float16*  o16     = (_Float16*)alloc((size_t)N * L0D * sizeof(_Float16));
  float*     egate   = (float*)alloc((size_t)E * NHEAD * sizeof(float));
  _Float16*  wk16    = (_Float16*)alloc((size_t)3 * L0D * L0D * sizeof(_Float16));
  _Float16*  wo16    = (_Float16*)alloc((size_t)L0D * L0D * sizeof(_Float16));
  _Float16*  w116    = (_Float16*)alloc((size_t)4 * L0D * L0D * sizeof(_Float16));

  offsets_kernel<<<1, 256, 0, stream>>>(eidx, N, offsets);
  copy_feats_kernel<<<(N * L0D + 255) / 256, 256, 0, stream>>>(features, feats, N);

  for (int layer = 0; layer < 2; ++layer) {
    int p = 2 + layer * 11;
    const float *ln_g = gf(p + 0), *ln_b = gf(p + 1);
    const float *Wkqv = gf(p + 2), *bkqv = gf(p + 3);
    const float *Wout = gf(p + 4), *bout = gf(p + 5);
    const float *bn_g = gf(p + 6), *bn_b = gf(p + 7);
    const float *Wb1  = gf(p + 8), *Wb2 = gf(p + 9), *bb2 = gf(p + 10);

    // pre-convert this layer's GEMM weights to f16 (tiny vs GEMM cost)
    cvt_f16_kernel<<<(3 * L0D * L0D + 255) / 256, 256, 0, stream>>>(Wkqv, wk16,
                                                                   3 * L0D * L0D);
    cvt_f16_kernel<<<(L0D * L0D + 255) / 256, 256, 0, stream>>>(Wout, wo16,
                                                                L0D * L0D);

    layernorm_kernel<<<(N + 7) / 8, 256, 0, stream>>>(feats, ln_g, ln_b, xln, N);

    int nt_kqv = ((N + 15) / 16) * (3 * L0D / 128);   // row tiles x col groups
    gemm_wmma_kernel<<<(nt_kqv + 7) / 8, 256, 0, stream>>>(
        xln, wk16, bkqv, nullptr, nullptr, kqv16, N, 3 * L0D);

    int emlp_blocks = (E + 255) / 256;
    if (emlp_blocks > 2048) emlp_blocks = 2048;
    edge_mlp_kernel<<<emlp_blocks, 256, 0, stream>>>(rbf, bn_g, bn_b, Wb1, Wb2,
                                                     bb2, egate, E);

    attn_kernel<<<NGRAPH * NHEAD, 256, ATTN_SMEM, stream>>>(kqv16, egate, edges,
                                                            E, offsets, o16);

    int nt_out = ((N + 15) / 16) * (L0D / 128);
    gemm_wmma_kernel<<<(nt_out + 7) / 8, 256, 0, stream>>>(
        o16, wo16, bout, feats, feats, nullptr, N, L0D);
  }

  // final FF head
  const float *ff_g = gf(24), *ff_b = gf(25), *W1 = gf(26), *W2 = gf(27),
              *b2 = gf(28);
  cvt_f16_kernel<<<(4 * L0D * L0D + 255) / 256, 256, 0, stream>>>(W1, w116,
                                                                 4 * L0D * L0D);
  layernorm_kernel<<<(N + 7) / 8, 256, 0, stream>>>(feats, ff_g, ff_b, xln, N);
  ff_kernel<<<(N + 15) / 16, 256, 0, stream>>>(xln, w116, W2, b2, out, N);
}
